// AtenMmQMixedSigni8_54752243089864
// MI455X (gfx1250) — 8-bit matrix operand order repaired and checked by reading, compiled, not run on hardware
//
#include <hip/hip_runtime.h>

typedef __attribute__((ext_vector_type(8))) int v8i;

#define MD 4096
#define ND 4096
#define KD 4096
#define KT (KD / 64)

__device__ __forceinline__ int swz16(int L, int half) {
  return (half ^ ((L >> 3) & 1)) * 16;
}

__device__ __forceinline__ void async_ld16(unsigned lds_off, const void* gaddr) {
  asm volatile("global_load_async_to_lds_b128 %0, %1, off"
               :: "v"(lds_off), "v"(gaddr) : "memory");
}
__device__ __forceinline__ void wait_async4() {
  asm volatile("s_wait_asynccnt 0x4" ::: "memory");
}
__device__ __forceinline__ void wait_async0() {
  asm volatile("s_wait_asynccnt 0x0" ::: "memory");
}

__global__ __launch_bounds__(256)
void zero_sums_kernel(int* __restrict__ sums) {
  sums[blockIdx.x * 256 + threadIdx.x] = 0;
}

__global__ __launch_bounds__(256)
void quantize_x_kernel(const float* __restrict__ x, char* __restrict__ xqp,
                       int* __restrict__ rowsum) {
  __shared__ unsigned char tile[16][264];
  __shared__ int red[256];
  const int mt = blockIdx.x;
  const int ktg = blockIdx.y;
  const int t = threadIdx.x;

  const int r = t >> 4;
  const int kk = (t & 15) * 16;
  const float* p = x + (size_t)(mt * 16 + r) * KD + ktg * 256 + kk;
  int s = 0;
#pragma unroll
  for (int i = 0; i < 4; ++i) {
    float4 v = ((const float4*)p)[i];
    int a0 = (int)v.x, a1 = (int)v.y, a2 = (int)v.z, a3 = (int)v.w;
    s += a0 + a1 + a2 + a3;
    tile[r][kk + i * 4 + 0] = (unsigned char)a0;
    tile[r][kk + i * 4 + 1] = (unsigned char)a1;
    tile[r][kk + i * 4 + 2] = (unsigned char)a2;
    tile[r][kk + i * 4 + 3] = (unsigned char)a3;
  }
  red[t] = s;
  __syncthreads();
  if ((t & 15) < 8) red[t] += red[t + 8];
  __syncthreads();
  if ((t & 15) < 4) red[t] += red[t + 4];
  __syncthreads();
  if ((t & 15) < 2) red[t] += red[t + 2];
  __syncthreads();
  if ((t & 15) == 0) atomicAdd(&rowsum[mt * 16 + r], red[t] + red[t + 1]);

  const int q = t >> 6;
  const int L = (t >> 1) & 31;
  const int half = t & 1;
  const int l16 = L & 15;
  const int kh = L >> 4;
  unsigned pk[4];
#pragma unroll
  for (int w = 0; w < 4; ++w) {
    unsigned v = 0;
#pragma unroll
    for (int c = 0; c < 4; ++c) {
      const int b = half * 16 + w * 4 + c;
      const int k_in = q * 64 + (b >> 3) * 16 + kh * 8 + (b & 7);
      v |= (unsigned)tile[l16][k_in] << (8 * c);
    }
    pk[w] = v;
  }
  const int kt = ktg * 4 + q;
  *(int4*)(xqp + ((size_t)(mt * KT + kt)) * 1024 + L * 32 + swz16(L, half)) =
      make_int4((int)pk[0], (int)pk[1], (int)pk[2], (int)pk[3]);
}

__global__ __launch_bounds__(256)
void quantize_y_kernel(const float* __restrict__ y, unsigned char* __restrict__ yqp,
                       int* __restrict__ colsum) {
  __shared__ unsigned char tile[64][68];
  __shared__ int colred[256];
  const int k0 = blockIdx.x * 64;
  const int n0 = blockIdx.y * 64;
  const int t = threadIdx.x;

  const int nl = t & 63;
  const int kb = t >> 6;
  int s = 0;
#pragma unroll
  for (int i = 0; i < 16; ++i) {
    const int kl = kb + i * 4;
    const int v = (int)y[(size_t)(k0 + kl) * ND + (n0 + nl)];
    s += v;
    tile[kl][nl] = (unsigned char)v;
  }
  colred[t] = s;
  __syncthreads();
  if (t < 64)
    atomicAdd(&colsum[n0 + t],
              colred[t] + colred[t + 64] + colred[t + 128] + colred[t + 192]);

  const int q = t >> 6;
  const int L = (t >> 1) & 31;
  const int half = t & 1;
  const int l16 = L & 15;
  const int kh = L >> 4;
  const int n_local = q * 16 + l16;
  const int kbase = half * 32 + kh * 8;
  unsigned pk[4];
#pragma unroll
  for (int w = 0; w < 4; ++w) {
    unsigned v = 0;
#pragma unroll
    for (int c = 0; c < 4; ++c)
      v |= (unsigned)tile[kbase + (w >> 1) * 16 + (w & 1) * 4 + c][n_local] << (8 * c);
    pk[w] = v;
  }
  const int nt = (n0 >> 4) + q;
  const int kt = k0 >> 6;
  *(int4*)(yqp + ((size_t)(nt * KT + kt)) * 1024 + L * 32 + swz16(L, half)) =
      make_int4((int)pk[0], (int)pk[1], (int)pk[2], (int)pk[3]);
}

extern "C" __global__ __launch_bounds__(512)
void gemm_iu8_wmma_kernel(const char* __restrict__ xqp,
                          const unsigned char* __restrict__ yqp,
                          const int* __restrict__ rowsum,
                          const int* __restrict__ colsum,
                          float* __restrict__ out) {
  extern __shared__ char smem[];
  const int t = threadIdx.x;
  const int lane = t & 31;
  const int wave = t >> 5;
  const int l16 = lane & 15;
  const int kh = lane >> 4;
  const int wm = wave & 3;
  const int wn = wave >> 2;

  const int m0 = blockIdx.x * 256 + wm * 64;
  const int n0 = blockIdx.y * 256 + wn * 64;

  const int tch = t & 63;
  const int ttile = t >> 6;
  const char* gA = xqp + ((size_t)(blockIdx.x * 16 + ttile) * KT) * 1024 + tch * 16;
  const char* gB = (const char*)yqp +
                   ((size_t)(blockIdx.y * 16 + ttile) * KT) * 1024 + tch * 16;
  const size_t tile_stride = (size_t)KT * 1024 * 8;
  const unsigned lds0 = (unsigned)(size_t)smem;
  const unsigned lA = lds0 + ttile * 1024 + tch * 16;
  const unsigned lB = lds0 + 16384 + ttile * 1024 + tch * 16;

  v8i acc[4][4];
#pragma unroll
  for (int f = 0; f < 4; ++f)
#pragma unroll
    for (int g = 0; g < 4; ++g)
#pragma unroll
      for (int j = 0; j < 8; ++j) acc[f][g][j] = 0;

#pragma unroll
  for (int pk = 0; pk < 2; ++pk) {
    const unsigned lb = pk * 32768u;
    const size_t go = (size_t)pk * 1024;
    async_ld16(lA + lb, gA + go);
    async_ld16(lA + lb + 8 * 1024, gA + go + tile_stride);
    async_ld16(lB + lb, gB + go);
    async_ld16(lB + lb + 8 * 1024, gB + go + tile_stride);
  }
  wait_async4();
  __syncthreads();

  const int s16 = swz16(lane, 0);
  const int s16b = 16 - s16;

  int bi = 0;
  for (int kt = 0; kt < KT; ++kt) {
    const unsigned buf = (unsigned)bi * 32768u;
    int si = bi + 2; if (si >= 3) si -= 3;
    const unsigned stg = (unsigned)si * 32768u;

    if (kt + 2 < KT) {
      const size_t go = (size_t)(kt + 2) * 1024;
      async_ld16(lA + stg, gA + go);
      async_ld16(lA + stg + 8 * 1024, gA + go + tile_stride);
      async_ld16(lB + stg, gB + go);
      async_ld16(lB + stg + 8 * 1024, gB + go + tile_stride);
    }

    v8i a[4], b[4];
#pragma unroll
    for (int f = 0; f < 4; ++f) {
      const char* p = smem + buf + (wm * 4 + f) * 1024 + lane * 32;
      int4 lo = *(const int4*)(p + s16);
      int4 hi = *(const int4*)(p + s16b);
      a[f][0] = lo.x; a[f][1] = lo.y; a[f][2] = lo.z; a[f][3] = lo.w;
      a[f][4] = hi.x; a[f][5] = hi.y; a[f][6] = hi.z; a[f][7] = hi.w;
    }
#pragma unroll
    for (int g = 0; g < 4; ++g) {
      const char* p = smem + buf + 16384 + (wn * 4 + g) * 1024 + lane * 32;
      int4 lo = *(const int4*)(p + s16);
      int4 hi = *(const int4*)(p + s16b);
      b[g][0] = lo.x; b[g][1] = lo.y; b[g][2] = lo.z; b[g][3] = lo.w;
      b[g][4] = hi.x; b[g][5] = hi.y; b[g][6] = hi.z; b[g][7] = hi.w;
    }
#pragma unroll
    for (int f = 0; f < 4; ++f)
#pragma unroll
      for (int g = 0; g < 4; ++g)
        acc[f][g] = __builtin_amdgcn_wmma_i32_16x16x64_iu8(
            true, a[f], false, b[g], acc[f][g], false, false);

    if (kt + 2 < KT) wait_async4(); else wait_async0();
    __syncthreads();

    if (++bi == 3) bi = 0;
  }

  const float S = 0.03f * 0.025f;
  const int CK = KD * (-66) * 160;
#pragma unroll
  for (int g = 0; g < 4; ++g) {
    const int n = n0 + g * 16 + l16;
    const int cs66 = 66 * colsum[n];
#pragma unroll
    for (int f = 0; f < 4; ++f) {
#pragma unroll
      for (int j = 0; j < 8; ++j) {
        const int m = m0 + f * 16 + kh * 8 + j;
        const int corr = cs66 - 160 * rowsum[m] + CK;
        out[(size_t)m * ND + n] = S * (float)(acc[f][g][j] + corr);
      }
    }
  }
}

extern "C" void kernel_launch(void* const* d_in, const int* in_sizes, int n_in,
                              void* d_out, int out_size, void* d_ws, size_t ws_size,
                              hipStream_t stream) {
  (void)in_sizes; (void)n_in; (void)out_size; (void)ws_size;
  const float* x = (const float*)d_in[0];
  const float* y = (const float*)d_in[1];
  float* out = (float*)d_out;

  char* xqp = (char*)d_ws;
  unsigned char* yqp = (unsigned char*)d_ws + (size_t)MD * KD;
  int* rowsum = (int*)((char*)d_ws + 2 * (size_t)MD * KD);
  int* colsum = rowsum + MD;

  zero_sums_kernel<<<(MD + ND) / 256, 256, 0, stream>>>(rowsum);
  quantize_x_kernel<<<dim3(MD / 16, KD / 256), 256, 0, stream>>>(x, xqp, rowsum);
  quantize_y_kernel<<<dim3(KD / 64, ND / 64), 256, 0, stream>>>(y, yqp, colsum);
  gemm_iu8_wmma_kernel<<<dim3(MD / 256, ND / 256), 512, 98304, stream>>>(
      xqp, yqp, rowsum, colsum, out);
}
